// BiDirectionalRNN_26963804685112
// MI455X (gfx1250) — matrix operand order repaired and checked by reading, not compiled, not run on hardware
//
#include <hip/hip_runtime.h>
#include <hip/hip_bf16.h>

typedef __attribute__((ext_vector_type(16))) __bf16 v16bf;
typedef __attribute__((ext_vector_type(8)))  __bf16 v8bf;
typedef __attribute__((ext_vector_type(4)))  __bf16 v4bf;
typedef __attribute__((ext_vector_type(8)))  float  v8f;
typedef __attribute__((ext_vector_type(4)))  float  v4f;

#define MODULUS 97
#define EMB     512
#define HID     512
#define TSTEPS  32
#define BATCH   1024
#define KBIG    (TSTEPS * 2 * HID)   // 32768
#define SLICES  8                    // split-K factor for the big GEMM

// GEMM tiling
#define BM 128
#define BN 128
#define BK 64
#define LPITCH 72                    // padded LDS pitch (elements) to spread banks

// ---------------- fp32 -> bf16 bulk convert (x4 vectorized) ----------------
__global__ __launch_bounds__(256) void cvt_f32_bf16_v4(const float* __restrict__ s,
                                                       __bf16* __restrict__ d, int n4) {
  const v4f* sv = (const v4f*)s;
  v4bf* dv = (v4bf*)d;
  int i = blockIdx.x * blockDim.x + threadIdx.x;
  int stride = gridDim.x * blockDim.x;
  for (; i < n4; i += stride) {
    v4f v = sv[i];
    v4bf o;
#pragma unroll
    for (int j = 0; j < 4; ++j) o[j] = (__bf16)v[j];
    dv[i] = o;
  }
}

// ---------------- table[c][h] = dot(emb[c,:], W[h,:])  (97 x 512, K=512) ----------------
__global__ __launch_bounds__(512) void table_gemm(const float* __restrict__ emb,
                                                  const float* __restrict__ W,
                                                  float* __restrict__ table) {
  __shared__ float e[EMB];
  const int c = blockIdx.x;
  const int h = threadIdx.x;
  e[h] = emb[c * EMB + h];
  __syncthreads();
  const v4f* wr = (const v4f*)(W + (size_t)h * EMB);
  const v4f* er = (const v4f*)e;
  float acc = 0.f;
#pragma unroll 4
  for (int d = 0; d < EMB / 4; ++d) {
    v4f w = wr[d], ev = er[d];
    acc += w[0] * ev[0] + w[1] * ev[1] + w[2] * ev[2] + w[3] * ev[3];
  }
  table[c * HID + h] = acc;
}

// ---------------- bidirectional RNN collapsed to elementwise scans ----------------
__global__ __launch_bounds__(512) void rnn_scan(const int* __restrict__ x,
                                                const float* __restrict__ tf,
                                                const float* __restrict__ tb,
                                                const float* __restrict__ bfv,
                                                const float* __restrict__ bbv,
                                                __bf16* __restrict__ Xcat) {
  __shared__ int xs[TSTEPS];
  const int b = blockIdx.x;
  const int h = threadIdx.x;
  if (h < TSTEPS) xs[h] = x[b * TSTEPS + h];
  __syncthreads();
  const float bfh = bfv[h];
  const float bbh = bbv[h];
  float hf = 0.f, hb = 0.f;
  __bf16* row = Xcat + (size_t)b * KBIG;
#pragma unroll
  for (int t = 0; t < TSTEPS; ++t) {
    hf = fmaxf(hf + tf[xs[t] * HID + h] + bfh, 0.f);
    row[t * (2 * HID) + h] = (__bf16)hf;
    hb = fmaxf(hb + tb[xs[TSTEPS - 1 - t] * HID + h] + bbh, 0.f);
    row[t * (2 * HID) + HID + h] = (__bf16)hb;
  }
}

// ---------------- WMMA fragment gather from LDS (two 16B halves) ----------------
__device__ __forceinline__ v16bf frag2(const __bf16* __restrict__ p, int o0, int o1) {
  v8bf lo = *(const v8bf*)(p + o0);
  v8bf hi = *(const v8bf*)(p + o1);
  return __builtin_shufflevector(lo, hi, 0, 1, 2, 3, 4, 5, 6, 7,
                                 8, 9, 10, 11, 12, 13, 14, 15);
}

// ---------------- Tiled NT GEMM: C/P = A[M,K] @ Bw[N,K]^T ----------------
// Block: 128M x 128N, 8 waves, wave = 32M x 64N (2 A-frags x 4 B-frags).
// LDS double-buffered BK=64 chunks; global loads for chunk i+1 overlap WMMAs of chunk i.
// SPLIT=true: write fp32 partials P[slice]; SPLIT=false: fused bias+ReLU -> bf16 C.
template <bool SPLIT>
__global__ __launch_bounds__(256) void gemm_wmma_tile(
    const __bf16* __restrict__ A, const __bf16* __restrict__ Bw,
    const float* __restrict__ bias, __bf16* __restrict__ C,
    float* __restrict__ P, int M, int N, int K, int kSlice) {
  __shared__ __align__(32) __bf16 smem[2 * 2 * BM * LPITCH];   // 72 KB
  const int t    = threadIdx.x;
  const int lane = t & 31;
  const int wave = t >> 5;
  const int l15  = lane & 15;
  const int hi   = lane >> 4;
  const int mSub = wave >> 1;      // 0..3
  const int nSub = wave & 1;       // 0..1
  const int mBlock = blockIdx.y * BM;
  const int nBlock = blockIdx.x * BN;
  const int kBegin = blockIdx.z * kSlice;
  const size_t sK = (size_t)K;

  // cooperative staging map: 1024 16B-chunks per matrix per BK-chunk, 4 per thread
  int rowIdx[4], off8[4];
#pragma unroll
  for (int p = 0; p < 4; ++p) {
    int cid = p * 256 + t;
    rowIdx[p] = cid >> 3;          // 0..127
    off8[p]  = (cid & 7) * 8;      // 0..56 elements
  }
  v8bf ra[4], rb[4];

  auto stage_load = [&](int kc) {
#pragma unroll
    for (int p = 0; p < 4; ++p) {
      ra[p] = *(const v8bf*)(A  + (size_t)(mBlock + rowIdx[p]) * sK + kc + off8[p]);
      rb[p] = *(const v8bf*)(Bw + (size_t)(nBlock + rowIdx[p]) * sK + kc + off8[p]);
    }
  };
  auto stage_store = [&](int buf) {
    __bf16* sA = smem + buf * (2 * BM * LPITCH);
    __bf16* sB = sA + BM * LPITCH;
#pragma unroll
    for (int p = 0; p < 4; ++p) {
      *(v8bf*)(sA + rowIdx[p] * LPITCH + off8[p]) = ra[p];
      *(v8bf*)(sB + rowIdx[p] * LPITCH + off8[p]) = rb[p];
    }
  };

  v8f zero = {};
  v8f acc[2][4];
#pragma unroll
  for (int aj = 0; aj < 2; ++aj)
#pragma unroll
    for (int j = 0; j < 4; ++j) acc[aj][j] = zero;

  stage_load(kBegin);
  stage_store(0);
  __syncthreads();

  const int nc = kSlice / BK;
  for (int i = 0; i < nc; ++i) {
    if (i + 1 < nc) stage_load(kBegin + (i + 1) * BK);   // overlap with WMMAs below
    const __bf16* sA = smem + (i & 1) * (2 * BM * LPITCH);
    const __bf16* sB = sA + BM * LPITCH;
#pragma unroll
    for (int kk = 0; kk < BK; kk += 32) {
      const __bf16* aBase = sA + (mSub * 32 + l15) * LPITCH + kk;
      v16bf a0 = frag2(aBase,               hi * 16, hi * 16 + 8);
      v16bf a1 = frag2(aBase + 16 * LPITCH, hi * 16, hi * 16 + 8);
      const __bf16* bBase = sB + (nSub * 64 + l15) * LPITCH + kk;
      v16bf b0 = frag2(bBase,               hi * 16, hi * 16 + 8);
      v16bf b1 = frag2(bBase + 16 * LPITCH, hi * 16, hi * 16 + 8);
      v16bf b2 = frag2(bBase + 32 * LPITCH, hi * 16, hi * 16 + 8);
      v16bf b3 = frag2(bBase + 48 * LPITCH, hi * 16, hi * 16 + 8);
      acc[0][0] = __builtin_amdgcn_wmma_f32_16x16x32_bf16(false, a0, false, b0, (short)0, acc[0][0], false, false);
      acc[0][1] = __builtin_amdgcn_wmma_f32_16x16x32_bf16(false, a0, false, b1, (short)0, acc[0][1], false, false);
      acc[0][2] = __builtin_amdgcn_wmma_f32_16x16x32_bf16(false, a0, false, b2, (short)0, acc[0][2], false, false);
      acc[0][3] = __builtin_amdgcn_wmma_f32_16x16x32_bf16(false, a0, false, b3, (short)0, acc[0][3], false, false);
      acc[1][0] = __builtin_amdgcn_wmma_f32_16x16x32_bf16(false, a1, false, b0, (short)0, acc[1][0], false, false);
      acc[1][1] = __builtin_amdgcn_wmma_f32_16x16x32_bf16(false, a1, false, b1, (short)0, acc[1][1], false, false);
      acc[1][2] = __builtin_amdgcn_wmma_f32_16x16x32_bf16(false, a1, false, b2, (short)0, acc[1][2], false, false);
      acc[1][3] = __builtin_amdgcn_wmma_f32_16x16x32_bf16(false, a1, false, b3, (short)0, acc[1][3], false, false);
    }
    if (i + 1 < nc) stage_store((i + 1) & 1);
    __syncthreads();
  }

  // epilogue: VGPR r -> row = base + hi*8 + r, col = base + l15
#pragma unroll
  for (int aj = 0; aj < 2; ++aj) {
#pragma unroll
    for (int j = 0; j < 4; ++j) {
      const int col  = nBlock + nSub * 64 + j * 16 + l15;
      const int row0 = mBlock + mSub * 32 + aj * 16 + hi * 8;
      if constexpr (SPLIT) {
        float* Pp = P + (size_t)blockIdx.z * M * N;
#pragma unroll
        for (int r = 0; r < 8; ++r) Pp[(size_t)(row0 + r) * N + col] = acc[aj][j][r];
      } else {
        const float bv = bias[col];
#pragma unroll
        for (int r = 0; r < 8; ++r) {
          float v = fmaxf(acc[aj][j][r] + bv, 0.f);
          C[(size_t)(row0 + r) * N + col] = (__bf16)v;
        }
      }
    }
  }
}

// ---------------- split-K reduce + bias + ReLU -> bf16 ----------------
__global__ __launch_bounds__(256) void splitk_fuse(const float* __restrict__ P,
                                                   const float* __restrict__ bias,
                                                   __bf16* __restrict__ C,
                                                   int MN, int N, int S) {
  const int MN4 = MN >> 2;
  const v4f* Pv = (const v4f*)P;
  int i = blockIdx.x * blockDim.x + threadIdx.x;
  int stride = gridDim.x * blockDim.x;
  for (; i < MN4; i += stride) {
    v4f s = Pv[i];
    for (int sl = 1; sl < S; ++sl) s += Pv[(size_t)sl * MN4 + i];
    v4f bv = *(const v4f*)(bias + ((i * 4) % N));
    v4bf o;
#pragma unroll
    for (int j = 0; j < 4; ++j) o[j] = (__bf16)fmaxf(s[j] + bv[j], 0.f);
    ((v4bf*)C)[i] = o;
  }
}

// ---------------- out[b][c] = dot(h[b,:], Wo[c,:]) + bo[c]  (fp32 head, N=97) ----------------
__global__ __launch_bounds__(128) void out_layer(const __bf16* __restrict__ H,
                                                 const float* __restrict__ Wo,
                                                 const float* __restrict__ bo,
                                                 float* __restrict__ out) {
  __shared__ float hs[HID];
  const int b = blockIdx.x;
  for (int i = threadIdx.x; i < HID; i += blockDim.x)
    hs[i] = (float)H[(size_t)b * HID + i];
  __syncthreads();
  const int c = threadIdx.x;
  if (c >= MODULUS) return;
  const v4f* wr = (const v4f*)(Wo + (size_t)c * HID);
  const v4f* hr = (const v4f*)hs;
  float acc = bo[c];
#pragma unroll 4
  for (int d = 0; d < HID / 4; ++d) {
    v4f w = wr[d], hv = hr[d];
    acc += w[0] * hv[0] + w[1] * hv[1] + w[2] * hv[2] + w[3] * hv[3];
  }
  out[b * MODULUS + c] = acc;
}

extern "C" void kernel_launch(void* const* d_in, const int* in_sizes, int n_in,
                              void* d_out, int out_size, void* d_ws, size_t ws_size,
                              hipStream_t stream) {
  (void)in_sizes; (void)n_in; (void)out_size; (void)ws_size;
  const int*   x   = (const int*)  d_in[0];
  const float* emb = (const float*)d_in[1];
  const float* Wf  = (const float*)d_in[2];
  const float* bfv = (const float*)d_in[3];
  const float* Wb  = (const float*)d_in[4];
  const float* bbv = (const float*)d_in[5];
  const float* W1  = (const float*)d_in[6];
  const float* b1  = (const float*)d_in[7];
  const float* W2  = (const float*)d_in[8];
  const float* b2  = (const float*)d_in[9];
  const float* Wo  = (const float*)d_in[10];
  const float* bo  = (const float*)d_in[11];
  float* out = (float*)d_out;

  // Workspace layout (all offsets 256B-aligned); total ~120.5 MB
  char* ws = (char*)d_ws;
  const size_t SZ_TABLE = (size_t)MODULUS * HID * sizeof(float);      // 198656
  const size_t SZ_W1B   = (size_t)HID * KBIG * sizeof(__bf16);        // 33554432
  const size_t SZ_W2B   = (size_t)HID * HID * sizeof(__bf16);         // 524288
  const size_t SZ_XCAT  = (size_t)BATCH * KBIG * sizeof(__bf16);      // 67108864
  const size_t SZ_H     = (size_t)BATCH * HID * sizeof(__bf16);       // 1048576
  float*  tableF = (float*) (ws);
  float*  tableB = (float*) (ws + SZ_TABLE);
  __bf16* W1b    = (__bf16*)(ws + 2 * SZ_TABLE);
  __bf16* W2b    = (__bf16*)(ws + 2 * SZ_TABLE + SZ_W1B);
  __bf16* Xcat   = (__bf16*)(ws + 2 * SZ_TABLE + SZ_W1B + SZ_W2B);
  __bf16* h1     = (__bf16*)(ws + 2 * SZ_TABLE + SZ_W1B + SZ_W2B + SZ_XCAT);
  __bf16* h2     = (__bf16*)(ws + 2 * SZ_TABLE + SZ_W1B + SZ_W2B + SZ_XCAT + SZ_H);
  float*  P      = (float*) (ws + 2 * SZ_TABLE + SZ_W1B + SZ_W2B + SZ_XCAT + 2 * SZ_H); // 16 MB

  // 1) 97-row pre-GEMM tables: collapses both RNN input GEMMs to gathers
  table_gemm<<<MODULUS, HID, 0, stream>>>(emb, Wf, tableF);
  table_gemm<<<MODULUS, HID, 0, stream>>>(emb, Wb, tableB);

  // 2) weight converts fp32 -> bf16
  cvt_f32_bf16_v4<<<4096, 256, 0, stream>>>(W1, W1b, (HID * KBIG) / 4);
  cvt_f32_bf16_v4<<<256,  256, 0, stream>>>(W2, W2b, (HID * HID) / 4);

  // 3) elementwise bidirectional scan -> Xcat [B, T*2H] in reference layout
  rnn_scan<<<BATCH, HID, 0, stream>>>(x, tableF, tableB, bfv, bbv, Xcat);

  // 4) h1 = relu(Xcat @ W1^T + b1): M=1024, N=512, K=32768, split-K=8 (256 blocks)
  dim3 gBig(HID / BN, BATCH / BM, SLICES);
  gemm_wmma_tile<true><<<gBig, 256, 0, stream>>>(Xcat, W1b, b1, nullptr, P,
                                                 BATCH, HID, KBIG, KBIG / SLICES);
  splitk_fuse<<<512, 256, 0, stream>>>(P, b1, h1, BATCH * HID, HID, SLICES);

  // 5) four W2 layers (K=512, direct fused epilogue), ping-pong h1 <-> h2 (ends in h1)
  dim3 gSm(HID / BN, BATCH / BM, 1);
  gemm_wmma_tile<false><<<gSm, 256, 0, stream>>>(h1, W2b, b2, h2, nullptr, BATCH, HID, HID, HID);
  gemm_wmma_tile<false><<<gSm, 256, 0, stream>>>(h2, W2b, b2, h1, nullptr, BATCH, HID, HID, HID);
  gemm_wmma_tile<false><<<gSm, 256, 0, stream>>>(h1, W2b, b2, h2, nullptr, BATCH, HID, HID, HID);
  gemm_wmma_tile<false><<<gSm, 256, 0, stream>>>(h2, W2b, b2, h1, nullptr, BATCH, HID, HID, HID);

  // 6) fp32 output head [1024, 97]
  out_layer<<<BATCH, 128, 0, stream>>>(h1, Wo, bo, out);
}
